// OldHeteroGNN_6468220748384
// MI455X (gfx1250) — matrix operand order repaired and checked by reading, not compiled, not run on hardware
//
#include <hip/hip_runtime.h>
#include <hip/hip_bf16.h>
#include <stdint.h>

#define HDIM 128
#define ACT_STRIDE 136          // padded row stride (halves) for LDS act tiles
#define ENC_NEGINF 0x007fffffu  // monotonic-uint encoding of -inf

typedef __attribute__((ext_vector_type(8)))  _Float16 v8h;
typedef __attribute__((ext_vector_type(16))) _Float16 v16h;
typedef __attribute__((ext_vector_type(8)))  float    v8f;

// ---- order-preserving float <-> uint mapping for atomicMax on u32 ----------
static __device__ __forceinline__ unsigned enc_f32(float f) {
  unsigned b = __float_as_uint(f);
  return (b & 0x80000000u) ? ~b : (b | 0x80000000u);
}
static __device__ __forceinline__ float dec_f32_or_zero(unsigned u) {
  // untouched slots stay ENC_NEGINF and must decode to 0 (torch_scatter fill)
  unsigned b = (u & 0x80000000u) ? (u ^ 0x80000000u) : ~u;
  float f = __uint_as_float(b);
  return (u == ENC_NEGINF) ? 0.0f : f;
}

// ---- WMMA fragment helpers (this code's contiguous K order, k=16h+s, shared by A and B) -------
// A: 16x32 f16. lane holds row M = lane&15. lanes<16: K={kc..kc+15}
//    lanes>=16: K={kc+16..kc+31}
static __device__ __forceinline__ v16h load_afrag(const _Float16* rowBase, int kc, int hiA) {
  const v8h a0 = *(const v8h*)(rowBase + kc + hiA);
  const v8h a1 = *(const v8h*)(rowBase + kc + 8 + hiA);
  v16h A;
#pragma unroll
  for (int i = 0; i < 8; ++i) { A[i] = a0[i]; A[i + 8] = a1[i]; }
  return A;
}

// Full 16x128 = [16 rows] x [128 cols] tile GEMM: acc[j] covers cols j*16..j*16+15.
// Bt is a transposed f16 weight matrix laid out [N=128][K=128] (rows contiguous),
// so a B fragment is 16 contiguous halves (32B) per lane.
static __device__ __forceinline__ void gemm16x128(const _Float16* aRow,
                                                  const _Float16* __restrict__ Bt,
                                                  const float* __restrict__ bias,
                                                  int lane, v8f acc[8]) {
  const int hiA = (lane & 16) ? 16 : 0;
  const int hiB = (lane & 16) ? 16 : 0;
  const int col = lane & 15;
#pragma unroll
  for (int j = 0; j < 8; ++j) {
    const int N = j * 16 + col;
    const float bv = bias[N];
    v8f c;
#pragma unroll
    for (int i = 0; i < 8; ++i) c[i] = bv;
#pragma unroll
    for (int kc = 0; kc < HDIM; kc += 32) {
      v16h A = load_afrag(aRow, kc, hiA);
      v16h B = *(const v16h*)(Bt + (size_t)N * HDIM + kc + hiB);
      c = __builtin_amdgcn_wmma_f32_16x16x32_f16(false, A, false, B, (short)0, c,
                                                 false, false);
    }
    acc[j] = c;
  }
}

// ---- prep: convert + transpose weights to f16 [N][K] -----------------------
__global__ void prep_weights(const float* __restrict__ eeW2,
                             const float* __restrict__ fxW1,
                             const float* __restrict__ fxW2,
                             _Float16* __restrict__ eeW2t,
                             _Float16* __restrict__ fxW1t,
                             _Float16* __restrict__ fxW2t) {
  const int M = HDIM * HDIM;
  int i = blockIdx.x * blockDim.x + threadIdx.x;
  if (i < 3 * M) {
    int t = i / M, r = i - t * M, k = r >> 7, n = r & (HDIM - 1);
    eeW2t[(size_t)t * M + n * HDIM + k] = (_Float16)eeW2[i];
  } else if (i < 12 * M) {
    int j = i - 3 * M;
    int t = j / M, r = j - t * M, k = r >> 7, n = r & (HDIM - 1);
    fxW1t[(size_t)t * M + n * HDIM + k] = (_Float16)fxW1[j];
  } else if (i < 21 * M) {
    int j = i - 12 * M;
    int t = j / M, r = j - t * M, k = r >> 7, n = r & (HDIM - 1);
    fxW2t[(size_t)t * M + n * HDIM + k] = (_Float16)fxW2[j];
  }
}

// ---- h_a = x_agent @ emb_W_a + emb_b_a (K=4, pure VALU) --------------------
__global__ void ha_init(const float* __restrict__ x, const float* __restrict__ W,
                        const float* __restrict__ b, float* __restrict__ h_a, int n) {
  int i = blockIdx.x * blockDim.x + threadIdx.x;
  if (i >= n) return;
  int nd = i >> 7, c = i & (HDIM - 1);
  const float* xr = x + (size_t)nd * 4;
  float s = b[c];
#pragma unroll
  for (int k = 0; k < 4; ++k) s += xr[k] * W[k * HDIM + c];
  h_a[i] = s;
}

__global__ void agg_init(unsigned* __restrict__ agg, int n) {
  int i = blockIdx.x * blockDim.x + threadIdx.x;
  if (i < n) agg[i] = ENC_NEGINF;
}

// ---- fused per-(layer, edge-type) kernel -----------------------------------
// stage0: act1 = relu(ea @ eeW1 + eeB1)           (K=4, VALU, -> LDS f16)
// G1    : eattr = act1 @ eeW2t + eeB2             (WMMA, -> LDS f16)
// G2    : hid   = relu(eattr @ fxW1t + fxB1)      (WMMA, -> LDS f16)
// G3    : msg   = hid @ fxW2t + fxB2              (WMMA)
// scatter-max(msg) into agg by dst (monotonic-u32 atomicMax)
__global__ void __launch_bounds__(128)
hetero_edge_layer(const float* __restrict__ ea,
                  const float* __restrict__ eeW1, const float* __restrict__ eeB1,
                  const _Float16* __restrict__ eeW2t, const float* __restrict__ eeB2,
                  const _Float16* __restrict__ fxW1t, const float* __restrict__ fxB1,
                  const _Float16* __restrict__ fxW2t, const float* __restrict__ fxB2,
                  const int* __restrict__ dst, unsigned* __restrict__ agg, int E) {
  __shared__ __attribute__((aligned(32))) _Float16 actA[4][16 * ACT_STRIDE];
  __shared__ __attribute__((aligned(32))) _Float16 actB[4][16 * ACT_STRIDE];
  const int tid = threadIdx.x;
  const int wave = tid >> 5;
  const int lane = tid & 31;
  const int tile = blockIdx.x * 64 + wave * 16;  // wave's first edge row

  _Float16* tA = actA[wave];
  _Float16* tB = actB[wave];

  // stage0: uniform trip count (64 iters/lane) -> EXEC stays all-1s
#pragma unroll 4
  for (int t = lane; t < 16 * HDIM; t += 32) {
    const int r = t >> 7, c = t & (HDIM - 1);
    int e = tile + r;
    e = (e < E) ? e : (E - 1);  // clamp tail rows (atomics guarded later)
    const float* er = ea + (size_t)e * 4;
    float s = eeB1[c];
#pragma unroll
    for (int k = 0; k < 4; ++k) s += er[k] * eeW1[k * HDIM + c];
    tA[r * ACT_STRIDE + c] = (_Float16)fmaxf(s, 0.0f);
  }
  __syncthreads();

  const _Float16* aRowA = tA + (lane & 15) * ACT_STRIDE;
  const _Float16* aRowB = tB + (lane & 15) * ACT_STRIDE;
  const int col = lane & 15;
  const int mBase = (lane & 16) ? 8 : 0;
  v8f acc[8];

  // G1: edge-embed second linear (no relu) -> actB
  gemm16x128(aRowA, eeW2t, eeB2, lane, acc);
#pragma unroll
  for (int j = 0; j < 8; ++j) {
    const int N = j * 16 + col;
#pragma unroll
    for (int i = 0; i < 8; ++i) tB[(i + mBase) * ACT_STRIDE + N] = (_Float16)acc[j][i];
  }
  __syncthreads();

  // G2: fx first linear + relu -> actA
  gemm16x128(aRowB, fxW1t, fxB1, lane, acc);
#pragma unroll
  for (int j = 0; j < 8; ++j) {
    const int N = j * 16 + col;
#pragma unroll
    for (int i = 0; i < 8; ++i)
      tA[(i + mBase) * ACT_STRIDE + N] = (_Float16)fmaxf(acc[j][i], 0.0f);
  }
  __syncthreads();

  // G3: fx second linear, then scatter-max (guards only AFTER all WMMAs)
  gemm16x128(aRowA, fxW2t, fxB2, lane, acc);
#pragma unroll
  for (int j = 0; j < 8; ++j) {
    const int N = j * 16 + col;
#pragma unroll
    for (int i = 0; i < 8; ++i) {
      const int e = tile + i + mBase;
      if (e < E) {
        const int d = dst[e];
        atomicMax(&agg[(size_t)d * HDIM + N], enc_f32(acc[j][i]));
      }
    }
  }
}

// ---- HeteroConv(aggr='max') across 3 edge types + residual; reset aggs -----
__global__ void combine_reset(float* __restrict__ h_a, unsigned* __restrict__ agg,
                              int n) {
  int i = blockIdx.x * blockDim.x + threadIdx.x;
  if (i >= n) return;
  float m = fmaxf(fmaxf(dec_f32_or_zero(agg[i]), dec_f32_or_zero(agg[i + n])),
                  dec_f32_or_zero(agg[i + 2 * n]));
  h_a[i] += m;
  agg[i] = ENC_NEGINF;
  agg[i + n] = ENC_NEGINF;
  agg[i + 2 * n] = ENC_NEGINF;
}

// ---- field head: one wave per agent ----------------------------------------
__global__ void __launch_bounds__(128)
field_head(const float* __restrict__ h_a, const float* __restrict__ action,
           const float* __restrict__ W1, const float* __restrict__ B1,
           const float* __restrict__ W2, const float* __restrict__ B2,
           float* __restrict__ out, int NA) {
  const int agent = blockIdx.x * (blockDim.x >> 5) + (threadIdx.x >> 5);
  const int lane = threadIdx.x & 31;
  if (agent >= NA) return;
  const float* hr = h_a + (size_t)agent * HDIM;
  const float a0 = action[agent * 3 + 0];
  const float a1 = action[agent * 3 + 1];
  const float a2 = action[agent * 3 + 2];
  float acc = 0.0f;
#pragma unroll
  for (int q = 0; q < 4; ++q) {
    const int c = lane + q * 32;
    float s = B1[c];
    for (int k = 0; k < HDIM; ++k) s += hr[k] * W1[(size_t)k * HDIM + c];
    s += a0 * W1[(size_t)HDIM * HDIM + c] + a1 * W1[(size_t)(HDIM + 1) * HDIM + c] +
         a2 * W1[(size_t)(HDIM + 2) * HDIM + c];
    s = fmaxf(s, 0.0f);
    acc += s * W2[c];
  }
#pragma unroll
  for (int off = 16; off > 0; off >>= 1) acc += __shfl_down(acc, off, 32);
  if (lane == 0) out[agent] = acc + B2[0];
}

extern "C" void kernel_launch(void* const* d_in, const int* in_sizes, int n_in,
                              void* d_out, int out_size, void* d_ws, size_t ws_size,
                              hipStream_t stream) {
  const float* x_agent = (const float*)d_in[1];
  const float* ea[3] = {(const float*)d_in[3], (const float*)d_in[4],
                        (const float*)d_in[5]};
  const float* action = (const float*)d_in[6];
  const int* dstp[3] = {(const int*)d_in[8], (const int*)d_in[10],
                        (const int*)d_in[12]};
  const float* embWa = (const float*)d_in[15];
  const float* embBa = (const float*)d_in[16];
  const float* eeW1 = (const float*)d_in[19];
  const float* eeB1 = (const float*)d_in[20];
  const float* eeW2 = (const float*)d_in[21];
  const float* eeB2 = (const float*)d_in[22];
  const float* fxW1 = (const float*)d_in[23];
  const float* fxB1 = (const float*)d_in[24];
  const float* fxW2 = (const float*)d_in[25];
  const float* fxB2 = (const float*)d_in[26];
  const float* fldW1 = (const float*)d_in[27];
  const float* fldB1 = (const float*)d_in[28];
  const float* fldW2 = (const float*)d_in[29];
  const float* fldB2 = (const float*)d_in[30];

  const int NA = in_sizes[1] / 4;
  const int Et[3] = {in_sizes[3] / 4, in_sizes[4] / 4, in_sizes[5] / 4};
  const int M = HDIM * HDIM;
  const int nAgg = NA * HDIM;

  uint8_t* w = (uint8_t*)d_ws;
  float* h_a = (float*)w;        w += (size_t)nAgg * 4;
  unsigned* agg = (unsigned*)w;  w += (size_t)3 * nAgg * 4;
  _Float16* eeW2t = (_Float16*)w; w += (size_t)3 * M * 2;
  _Float16* fxW1t = (_Float16*)w; w += (size_t)9 * M * 2;
  _Float16* fxW2t = (_Float16*)w; w += (size_t)9 * M * 2;

  prep_weights<<<(21 * M + 255) / 256, 256, 0, stream>>>(eeW2, fxW1, fxW2, eeW2t,
                                                         fxW1t, fxW2t);
  ha_init<<<(nAgg + 255) / 256, 256, 0, stream>>>(x_agent, embWa, embBa, h_a, nAgg);
  agg_init<<<(3 * nAgg + 255) / 256, 256, 0, stream>>>(agg, 3 * nAgg);

  for (int l = 0; l < 3; ++l) {
    for (int t = 0; t < 3; ++t) {
      const int lt = l * 3 + t;
      const int grid = (Et[t] + 63) / 64;
      hetero_edge_layer<<<grid, 128, 0, stream>>>(
          ea[t], eeW1 + (size_t)t * 4 * HDIM, eeB1 + (size_t)t * HDIM,
          eeW2t + (size_t)t * M, eeB2 + (size_t)t * HDIM,
          fxW1t + (size_t)lt * M, fxB1 + (size_t)lt * HDIM,
          fxW2t + (size_t)lt * M, fxB2 + (size_t)lt * HDIM, dstp[t],
          agg + (size_t)t * nAgg, Et[t]);
    }
    combine_reset<<<(nAgg + 255) / 256, 256, 0, stream>>>(h_a, agg, nAgg);
  }
  field_head<<<(NA + 3) / 4, 128, 0, stream>>>(h_a, action, fldW1, fldB1, fldW2,
                                               fldB2, (float*)d_out, NA);
  (void)n_in; (void)out_size; (void)ws_size;
}
